// PreLNBlock_81054622810715
// MI455X (gfx1250) — matrix operand order repaired and checked by reading, not compiled, not run on hardware
//
#include <hip/hip_runtime.h>
#include <hip/hip_bf16.h>
#include <math.h>

// ---------------------------------------------------------------------------
// Pre-LN transformer block for MI455X (gfx1250).
// ~240 GFLOP vs ~0.2 GB traffic -> compute bound -> bf16 WMMA for every GEMM,
// TDM (tensor_load_to_lds) + LDS double buffering to feed the MMA pipes.
// ---------------------------------------------------------------------------

typedef __attribute__((ext_vector_type(16))) __bf16 v16bf;
typedef __attribute__((ext_vector_type(8)))  __bf16 v8bf;
typedef __attribute__((ext_vector_type(8)))  float  v8f;
typedef __attribute__((ext_vector_type(4)))  unsigned int u32x4;
typedef __attribute__((ext_vector_type(8)))  int i32x8;
typedef __attribute__((ext_vector_type(4)))  int i32x4;

#if defined(__has_builtin)
#if __has_builtin(__builtin_amdgcn_tensor_load_to_lds)
#define USE_TDM 1
#endif
#endif
#ifndef USE_TDM
#define USE_TDM 0
#endif

static __device__ __forceinline__ __bf16 f2bf(float f) {
    unsigned u = __builtin_bit_cast(unsigned, f);
    u += 0x7FFFu + ((u >> 16) & 1u);              // round-to-nearest-even
    unsigned short h = (unsigned short)(u >> 16);
    return __builtin_bit_cast(__bf16, h);
}

static __device__ __forceinline__ v16bf mk16(v8bf lo, v8bf hi) {
    v16bf r;
#pragma unroll
    for (int i = 0; i < 8; ++i) { r[i] = lo[i]; r[i + 8] = hi[i]; }
    return r;
}

static __device__ __forceinline__ v8f wmma_bf16(v16bf a, v16bf b, v8f c) {
    return __builtin_amdgcn_wmma_f32_16x16x32_bf16(
        false, a, false, b, (short)0, c, false, false);
}

#if USE_TDM
// Issue one TDM 2D tile load: tile_rows=128, tile_cols=32 bf16 elements,
// source row stride = stride_elems, destination LDS rows packed (64B/row).
// D# bitfields per CDNA5 ISA ch.8 (group0: count/lds/global/type,
// group1: data_size, tensor dims, tile dims, dim0 stride).
// This toolchain exposes the 6-arg builtin:
//   (uint32x4 g0, int32x8 g1, int32x4 g2, int32x4 g3, int32x8 extra, i32 cpol)
static __device__ __forceinline__ void tdm_load_tile(unsigned lds_off,
                                                     const __bf16* g,
                                                     int stride_elems) {
    unsigned long long ga = (unsigned long long)(size_t)g;
    u32x4 g0;
    g0.x = 1u;                                              // count=1 (user D#)
    g0.y = lds_off;                                         // lds_addr (bytes)
    g0.z = (unsigned)ga;                                    // global_addr[31:0]
    g0.w = (unsigned)((ga >> 32) & 0x01FFFFFFu) | 0x80000000u; // addr[56:32] | type=2
    i32x8 g1;
    g1[0] = 1 << 16;                                        // data_size=1 -> 2 bytes
    g1[1] = (stride_elems & 0xffff) << 16;                  // tensor_dim0 lo16
    g1[2] = (int)(((unsigned)stride_elems >> 16) & 0xffffu);// tensor_dim0 hi16, dim1 lo16=0
    g1[3] = 0x10 | (32 << 16);                              // tensor_dim1 hi16 (=1M rows), tile_dim0=32
    g1[4] = 128;                                            // tile_dim1=128, tile_dim2=0
    g1[5] = stride_elems;                                   // tensor_dim0_stride lo32
    g1[6] = 0;                                              // stride hi16 / dim1_stride lo16
    g1[7] = 0;
    i32x4 z4 = {0, 0, 0, 0};
    i32x8 z8 = {0, 0, 0, 0, 0, 0, 0, 0};
    __builtin_amdgcn_tensor_load_to_lds(g0, g1, z4, z4, z8, 0);
}
#endif

// ---------------------------------------------------------------------------
// fp32 -> bf16 conversion (weights)
// ---------------------------------------------------------------------------
__global__ __launch_bounds__(256) void cvt_bf16_kernel(const float* __restrict__ in,
                                                       __bf16* __restrict__ out, int n) {
    int i = blockIdx.x * 256 + threadIdx.x;
    if (i < n) out[i] = f2bf(in[i]);
}

// ---------------------------------------------------------------------------
// Pack bool mask bytes into 32-bit words via wave32 ballot.
// ---------------------------------------------------------------------------
__global__ __launch_bounds__(256) void mask_pack_kernel(const unsigned char* __restrict__ m,
                                                        unsigned* __restrict__ bits,
                                                        long long n) {
    long long i = (long long)blockIdx.x * 256 + threadIdx.x;
    unsigned char v = (i < n) ? m[i] : (unsigned char)0;
    unsigned w = (unsigned)__ballot(v != 0);
    if ((threadIdx.x & 31) == 0 && i < n) bits[i >> 5] = w;
}

// ---------------------------------------------------------------------------
// LayerNorm (fp32 in) -> bf16 out. One 256-thread block per row of D=1024.
// ---------------------------------------------------------------------------
__global__ __launch_bounds__(256) void ln_bf16_kernel(const float* __restrict__ x,
                                                      const float* __restrict__ s,
                                                      const float* __restrict__ b,
                                                      __bf16* __restrict__ out, int D) {
    const int row = blockIdx.x;
    const float* xr = x + (size_t)row * D;
    float lsum = 0.f, lsq = 0.f;
    for (int i = threadIdx.x; i < D; i += 256) {
        float v = xr[i];
        lsum += v; lsq += v * v;
    }
#pragma unroll
    for (int off = 16; off; off >>= 1) {
        lsum += __shfl_xor(lsum, off, 32);
        lsq  += __shfl_xor(lsq,  off, 32);
    }
    __shared__ float ssum[8], ssq[8];
    if ((threadIdx.x & 31) == 0) { ssum[threadIdx.x >> 5] = lsum; ssq[threadIdx.x >> 5] = lsq; }
    __syncthreads();
    float tsum = 0.f, tsq = 0.f;
#pragma unroll
    for (int i = 0; i < 8; ++i) { tsum += ssum[i]; tsq += ssq[i]; }
    const float mu  = tsum / (float)D;
    const float var = tsq / (float)D - mu * mu;
    const float inv = rsqrtf(var + 1e-5f);
    for (int i = threadIdx.x; i < D; i += 256) {
        float v = (xr[i] - mu) * inv * s[i] + b[i];
        out[(size_t)row * D + i] = f2bf(v);
    }
}

// ---------------------------------------------------------------------------
// bf16 WMMA GEMM:  C[M,N] = A[M,K] @ W[N,K]^T  (row-major, K contiguous)
// 256 threads = 8 waves (2 x 4). Block tile 128x128, wave tile 64x32, KT=32.
// A/B 128x32 tiles staged into LDS by the Tensor Data Mover (double buffered,
// TDM(i+1) overlaps compute(i)); fragments read back with ds_load_b128.
// Epilogue MODE: 0 Q/K scatter [B,H,L,dk] | 1 V^T scatter [B,H,dk,L]
//                2 +resid fp32 | 3 GELU bf16 | 4 +resid fp32 (final out)
// ---------------------------------------------------------------------------
template <int MODE>
__global__ __launch_bounds__(256) void gemm_bf16_kernel(
    const __bf16* __restrict__ A, const __bf16* __restrict__ W,
    int M, int N, int K,
    const float* __restrict__ bias,
    const float* __restrict__ resid,
    float*  __restrict__ outf,
    __bf16* __restrict__ outb) {
    const int lane = threadIdx.x & 31;
    const int wid  = threadIdx.x >> 5;
    const int wm   = wid >> 2;          // 0..1
    const int wn   = wid & 3;           // 0..3
    const int l15  = lane & 15;
    const int half = lane >> 4;
    const int bm = blockIdx.x * 128;
    const int bn = blockIdx.y * 128;

    __shared__ __align__(16) __bf16 smA[2][128 * 32];
    __shared__ __align__(16) __bf16 smB[2][128 * 32];

    v8f acc[4][2];
#pragma unroll
    for (int ms = 0; ms < 4; ++ms)
#pragma unroll
        for (int ns = 0; ns < 2; ++ns) acc[ms][ns] = (v8f){0.f,0.f,0.f,0.f,0.f,0.f,0.f,0.f};

    const int nK = K >> 5;

#if USE_TDM
    if (wid == 0) {
        tdm_load_tile((unsigned)(size_t)&smA[0][0], A + (size_t)bm * K, K);
        tdm_load_tile((unsigned)(size_t)&smB[0][0], W + (size_t)bn * K, K);
        __builtin_amdgcn_s_wait_tensorcnt(0);
    }
    __syncthreads();
    for (int ki = 0; ki < nK; ++ki) {
        const int cur = ki & 1, nxt = cur ^ 1;
        if ((ki + 1 < nK) && wid == 0) {   // overlap: TDM(i+1) during compute(i)
            tdm_load_tile((unsigned)(size_t)&smA[nxt][0],
                          A + (size_t)bm * K + (size_t)(ki + 1) * 32, K);
            tdm_load_tile((unsigned)(size_t)&smB[nxt][0],
                          W + (size_t)bn * K + (size_t)(ki + 1) * 32, K);
        }
        v16bf afrag[4], bfrag[2];
#pragma unroll
        for (int ms = 0; ms < 4; ++ms) {
            const __bf16* p = &smA[cur][(wm * 64 + ms * 16 + l15) * 32 + half * 8];
            afrag[ms] = mk16(*(const v8bf*)p, *(const v8bf*)(p + 16));
        }
#pragma unroll
        for (int ns = 0; ns < 2; ++ns) {
            const __bf16* p = &smB[cur][(wn * 32 + ns * 16 + l15) * 32 + half * 8];
            bfrag[ns] = mk16(*(const v8bf*)p, *(const v8bf*)(p + 16));
        }
#pragma unroll
        for (int ms = 0; ms < 4; ++ms)
#pragma unroll
            for (int ns = 0; ns < 2; ++ns)
                acc[ms][ns] = wmma_bf16(afrag[ms], bfrag[ns], acc[ms][ns]);
        __syncthreads();                    // all waves done with buffer `cur`
        if (wid == 0) __builtin_amdgcn_s_wait_tensorcnt(0);
        __syncthreads();                    // publish buffer `nxt`
    }
#else
    // Fallback: cooperative global->LDS staging (single buffer).
    for (int ki = 0; ki < nK; ++ki) {
        {
            const int t = (int)threadIdx.x;
            const int r = t >> 1, c = (t & 1) * 16;
            const __bf16* ga = A + (size_t)(bm + r) * K + ki * 32 + c;
            *(v8bf*)&smA[0][r * 32 + c]     = *(const v8bf*)ga;
            *(v8bf*)&smA[0][r * 32 + c + 8] = *(const v8bf*)(ga + 8);
            const __bf16* gb = W + (size_t)(bn + r) * K + ki * 32 + c;
            *(v8bf*)&smB[0][r * 32 + c]     = *(const v8bf*)gb;
            *(v8bf*)&smB[0][r * 32 + c + 8] = *(const v8bf*)(gb + 8);
        }
        __syncthreads();
        v16bf afrag[4], bfrag[2];
#pragma unroll
        for (int ms = 0; ms < 4; ++ms) {
            const __bf16* p = &smA[0][(wm * 64 + ms * 16 + l15) * 32 + half * 8];
            afrag[ms] = mk16(*(const v8bf*)p, *(const v8bf*)(p + 16));
        }
#pragma unroll
        for (int ns = 0; ns < 2; ++ns) {
            const __bf16* p = &smB[0][(wn * 32 + ns * 16 + l15) * 32 + half * 8];
            bfrag[ns] = mk16(*(const v8bf*)p, *(const v8bf*)(p + 16));
        }
#pragma unroll
        for (int ms = 0; ms < 4; ++ms)
#pragma unroll
            for (int ns = 0; ns < 2; ++ns)
                acc[ms][ns] = wmma_bf16(afrag[ms], bfrag[ns], acc[ms][ns]);
        __syncthreads();
    }
#endif

    // ---- epilogue (C layout: VGPR j -> row = 8*half + j, col = lane&15) ----
#pragma unroll
    for (int ms = 0; ms < 4; ++ms) {
#pragma unroll
        for (int ns = 0; ns < 2; ++ns) {
            const int col = bn + wn * 32 + ns * 16 + l15;
            const float bv = bias[col];
#pragma unroll
            for (int j = 0; j < 8; ++j) {
                const int row = bm + wm * 64 + ms * 16 + half * 8 + j;
                float v = acc[ms][ns][j] + bv;
                if (MODE == 0) {               // [B,H,L,dk]
                    const int bb = row >> 10, ll = row & 1023;
                    const int hh = col >> 6,  dd = col & 63;
                    outb[(((size_t)bb * 16 + hh) * 1024 + ll) * 64 + dd] = f2bf(v);
                } else if (MODE == 1) {        // V^T: [B,H,dk,L]
                    const int bb = row >> 10, ll = row & 1023;
                    const int hh = col >> 6,  dd = col & 63;
                    outb[(((size_t)bb * 16 + hh) * 64 + dd) * 1024 + ll] = f2bf(v);
                } else if (MODE == 2) {
                    const size_t idx = (size_t)row * N + col;
                    outf[idx] = v + resid[idx];
                } else if (MODE == 3) {
                    const float g = 0.5f * v * (1.0f + erff(v * 0.70710678118654752f));
                    outb[(size_t)row * N + col] = f2bf(g);
                } else {                       // MODE == 4
                    const size_t idx = (size_t)row * N + col;
                    outf[idx] = v + resid[idx];
                }
            }
        }
    }
}

// ---------------------------------------------------------------------------
// Flash-style attention. Grid (B*H, L/64); 128 threads = 4 waves, each wave
// owns 16 q rows, kv tiles of 32. Mask comes pre-packed as bits (1 = masked).
// ---------------------------------------------------------------------------
__global__ __launch_bounds__(128) void attn_kernel(
    const __bf16* __restrict__ Q,   // [B,H,L,64]
    const __bf16* __restrict__ Km,  // [B,H,L,64]
    const __bf16* __restrict__ VT,  // [B,H,64,L]
    const unsigned* __restrict__ mbits, // [B,L,L/32]
    __bf16* __restrict__ ctx,       // [B,L,D]
    int L, int H) {
    const int bh   = blockIdx.x;
    const int b    = bh / H;
    const int h    = bh % H;
    const int wid  = threadIdx.x >> 5;
    const int lane = threadIdx.x & 31;
    const int l15  = lane & 15;
    const int hf   = lane >> 4;
    const int q0   = blockIdx.y * 64 + wid * 16;
    const int LW   = L >> 5;

    const __bf16* Qb = Q  + (size_t)bh * L * 64;
    const __bf16* Kb = Km + (size_t)bh * L * 64;
    const __bf16* Vb = VT + (size_t)bh * 64 * L;
    const unsigned* mwb = mbits + (size_t)b * L * LW;

    __shared__ __align__(16) __bf16 pbuf[4][16 * 32];
    __bf16* pb = &pbuf[wid][0];

    v16bf qf[2];
    {
        const __bf16* p = Qb + (size_t)(q0 + l15) * 64 + hf * 8;
        qf[0] = mk16(*(const v8bf*)p,        *(const v8bf*)(p + 16));
        qf[1] = mk16(*(const v8bf*)(p + 32), *(const v8bf*)(p + 48));
    }

    const float NEGINF = -__builtin_inff();
    float m_i[8], l_i[8];
#pragma unroll
    for (int j = 0; j < 8; ++j) { m_i[j] = NEGINF; l_i[j] = 0.f; }
    v8f o[4];
#pragma unroll
    for (int dn = 0; dn < 4; ++dn) o[dn] = (v8f){0.f,0.f,0.f,0.f,0.f,0.f,0.f,0.f};

    for (int kt = 0; kt < L; kt += 32) {
        // mask words for this wave's 8 rows (replicated across the 16-lane half)
        unsigned mw[8];
#pragma unroll
        for (int j = 0; j < 8; ++j)
            mw[j] = mwb[(size_t)(q0 + hf * 8 + j) * LW + (kt >> 5)];

        // ---- S = Q K^T for 16x32 tile ----
        v8f s[2];
#pragma unroll
        for (int nt = 0; nt < 2; ++nt) {
            s[nt] = (v8f){0.f,0.f,0.f,0.f,0.f,0.f,0.f,0.f};
            const __bf16* kp = Kb + (size_t)(kt + nt * 16 + l15) * 64;
            v16bf b0 = mk16(*(const v8bf*)(kp + hf * 8),      *(const v8bf*)(kp + hf * 8 + 16));
            v16bf b1 = mk16(*(const v8bf*)(kp + 32 + hf * 8), *(const v8bf*)(kp + 32 + hf * 8 + 16));
            s[nt] = wmma_bf16(qf[0], b0, s[nt]);
            s[nt] = wmma_bf16(qf[1], b1, s[nt]);
        }
        // ---- scale + mask (bit test) ----
#pragma unroll
        for (int nt = 0; nt < 2; ++nt) {
#pragma unroll
            for (int j = 0; j < 8; ++j) {
                float v = s[nt][j] * 0.125f;
                if ((mw[j] >> (nt * 16 + l15)) & 1u) v = NEGINF;
                s[nt][j] = v;
            }
        }
        // ---- online softmax ----
        float corr[8];
#pragma unroll
        for (int j = 0; j < 8; ++j) {
            float t = fmaxf(s[0][j], s[1][j]);
#pragma unroll
            for (int off = 8; off; off >>= 1) t = fmaxf(t, __shfl_xor(t, off, 32));
            const float mn = fmaxf(m_i[j], t);
            const bool  dead = (mn == NEGINF);
            const float c  = dead ? 1.f : __expf(m_i[j] - mn);
            float p0 = dead ? 0.f : __expf(s[0][j] - mn);
            float p1 = dead ? 0.f : __expf(s[1][j] - mn);
            float ps = p0 + p1;
#pragma unroll
            for (int off = 8; off; off >>= 1) ps += __shfl_xor(ps, off, 32);
            l_i[j] = l_i[j] * c + ps;
            m_i[j] = mn;
            corr[j] = c;
            s[0][j] = p0; s[1][j] = p1;
        }
#pragma unroll
        for (int dn = 0; dn < 4; ++dn)
#pragma unroll
            for (int j = 0; j < 8; ++j) o[dn][j] *= corr[j];

        // ---- P (C layout) -> LDS -> A fragment (16x32) ----
#pragma unroll
        for (int nt = 0; nt < 2; ++nt)
#pragma unroll
            for (int j = 0; j < 8; ++j)
                pb[(hf * 8 + j) * 32 + nt * 16 + l15] = f2bf(s[nt][j]);
        asm volatile("s_wait_dscnt 0" ::: "memory");
        const __bf16* pr = pb + l15 * 32 + hf * 8;
        v16bf pa = mk16(*(const v8bf*)pr, *(const v8bf*)(pr + 16));

        // ---- O += P V (contiguous B fragments from V^T) ----
#pragma unroll
        for (int dn = 0; dn < 4; ++dn) {
            const __bf16* vp = Vb + (size_t)(dn * 16 + l15) * L + kt + hf * 8;
            v16bf vf = mk16(*(const v8bf*)vp, *(const v8bf*)(vp + 16));
            o[dn] = wmma_bf16(pa, vf, o[dn]);
        }
    }

    // ---- normalize + store bf16 to ctx [B,L,D] ----
#pragma unroll
    for (int dn = 0; dn < 4; ++dn) {
#pragma unroll
        for (int j = 0; j < 8; ++j) {
            const int qr = q0 + hf * 8 + j;
            const float den = l_i[j];
            const float v = (den > 0.f) ? o[dn][j] / den : 0.f;
            ctx[((size_t)b * L + qr) * 1024 + h * 64 + dn * 16 + l15] = f2bf(v);
        }
    }
}

// ---------------------------------------------------------------------------
// Host orchestration
// ---------------------------------------------------------------------------
extern "C" void kernel_launch(void* const* d_in, const int* in_sizes, int n_in,
                              void* d_out, int out_size, void* d_ws, size_t ws_size,
                              hipStream_t stream) {
    constexpr int B = 8, L = 1024, D = 1024, H = 16, F = 4096;
    constexpr int M = B * L;  // 8192 tokens

    const float* x    = (const float*)d_in[0];
    const unsigned char* mask = (const unsigned char*)d_in[1];
    const float* Wq   = (const float*)d_in[2];
    const float* bq   = (const float*)d_in[3];
    const float* Wk   = (const float*)d_in[4];
    const float* bk   = (const float*)d_in[5];
    const float* Wv   = (const float*)d_in[6];
    const float* bv   = (const float*)d_in[7];
    const float* Wo   = (const float*)d_in[8];
    const float* bo   = (const float*)d_in[9];
    const float* ln1s = (const float*)d_in[10];
    const float* ln1b = (const float*)d_in[11];
    const float* ln2s = (const float*)d_in[12];
    const float* ln2b = (const float*)d_in[13];
    const float* W1   = (const float*)d_in[14];
    const float* b1   = (const float*)d_in[15];
    const float* W2   = (const float*)d_in[16];
    const float* b2   = (const float*)d_in[17];
    float* out = (float*)d_out;

    char* ws = (char*)d_ws;
    size_t off = 0;
    auto carve = [&](size_t bytes) { void* p = ws + off; off += (bytes + 255) & ~(size_t)255; return p; };
    __bf16*   Wq_b  = (__bf16*)carve((size_t)D * D * 2);
    __bf16*   Wk_b  = (__bf16*)carve((size_t)D * D * 2);
    __bf16*   Wv_b  = (__bf16*)carve((size_t)D * D * 2);
    __bf16*   Wo_b  = (__bf16*)carve((size_t)D * D * 2);
    __bf16*   W1_b  = (__bf16*)carve((size_t)F * D * 2);
    __bf16*   W2_b  = (__bf16*)carve((size_t)D * F * 2);
    __bf16*   xn_b  = (__bf16*)carve((size_t)M * D * 2);
    __bf16*   Q_b   = (__bf16*)carve((size_t)M * D * 2);
    __bf16*   K_b   = (__bf16*)carve((size_t)M * D * 2);
    __bf16*   VT_b  = (__bf16*)carve((size_t)M * D * 2);
    __bf16*   ctx_b = (__bf16*)carve((size_t)M * D * 2);
    float*    x1    = (float*) carve((size_t)M * D * 4);
    __bf16*   xn2_b = (__bf16*)carve((size_t)M * D * 2);
    __bf16*   h_b   = (__bf16*)carve((size_t)M * F * 2);
    unsigned* mbits = (unsigned*)carve((size_t)B * L * (L / 32) * 4);
    (void)ws_size; (void)in_sizes; (void)n_in; (void)out_size;

    auto cvt = [&](const float* src, __bf16* dst, int n) {
        cvt_bf16_kernel<<<(n + 255) / 256, 256, 0, stream>>>(src, dst, n);
    };
    cvt(Wq, Wq_b, D * D);
    cvt(Wk, Wk_b, D * D);
    cvt(Wv, Wv_b, D * D);
    cvt(Wo, Wo_b, D * D);
    cvt(W1, W1_b, F * D);
    cvt(W2, W2_b, D * F);

    // pack bool mask to bits
    {
        long long n = (long long)B * L * L;
        mask_pack_kernel<<<(unsigned)((n + 255) / 256), 256, 0, stream>>>(mask, mbits, n);
    }

    // LN1 -> bf16 activations
    ln_bf16_kernel<<<M, 256, 0, stream>>>(x, ln1s, ln1b, xn_b, D);

    // Q,K (scatter [B,H,L,dk]), V (scatter to V^T [B,H,dk,L])
    dim3 gQ(M / 128, D / 128);
    gemm_bf16_kernel<0><<<gQ, 256, 0, stream>>>(xn_b, Wq_b, M, D, D, bq, nullptr, nullptr, Q_b);
    gemm_bf16_kernel<0><<<gQ, 256, 0, stream>>>(xn_b, Wk_b, M, D, D, bk, nullptr, nullptr, K_b);
    gemm_bf16_kernel<1><<<gQ, 256, 0, stream>>>(xn_b, Wv_b, M, D, D, bv, nullptr, nullptr, VT_b);

    // attention -> ctx bf16 [B,L,D]
    dim3 gA(B * H, L / 64);
    attn_kernel<<<gA, 128, 0, stream>>>(Q_b, K_b, VT_b, mbits, ctx_b, L, H);

    // output projection + residual -> x1 (fp32)
    gemm_bf16_kernel<2><<<gQ, 256, 0, stream>>>(ctx_b, Wo_b, M, D, D, bo, x, x1, nullptr);

    // LN2 -> bf16
    ln_bf16_kernel<<<M, 256, 0, stream>>>(x1, ln2s, ln2b, xn2_b, D);

    // FFN1 + exact GELU -> h bf16
    dim3 gF1(M / 128, F / 128);
    gemm_bf16_kernel<3><<<gF1, 256, 0, stream>>>(xn2_b, W1_b, M, F, D, b1, nullptr, nullptr, h_b);

    // FFN2 + residual -> d_out (fp32)
    dim3 gF2(M / 128, D / 128);
    gemm_bf16_kernel<4><<<gF2, 256, 0, stream>>>(h_b, W2_b, M, D, F, b2, x1, out, nullptr);
}
